// FrontDetector_46626164965539
// MI455X (gfx1250) — compile-verified, matrix operand order checked by reading, not run on hardware
//
#include <hip/hip_runtime.h>
#include <math.h>

typedef __attribute__((ext_vector_type(4)))  float    f32x4;
typedef __attribute__((ext_vector_type(8)))  float    v8f;
typedef __attribute__((ext_vector_type(16))) _Float16 v16h;
typedef __attribute__((ext_vector_type(8)))  _Float16 v8h;

#define XN   262144
#define NB   64
#define HH   256
#define KK   8
#define THRF 1e-6f

// ---------------------------------------------------------------------------
// Kernel A: stream density, count discontinuities, record first-8 per chunk.
// grid = (8 chunks, 64 rows), 256 threads; each thread owns 128 contiguous pairs.
// Memory-bound: 64 MB streamed once with B128 non-temporal loads.
// ---------------------------------------------------------------------------
__global__ __launch_bounds__(256) void scan_fronts(const float* __restrict__ x,
                                                   int* __restrict__ cntpart,
                                                   int* __restrict__ cand) {
  const int row   = blockIdx.y;
  const int chunk = blockIdx.x;
  const int tid   = threadIdx.x;
  const int S     = 128;
  const long base = (long)chunk * 256 * S + (long)tid * S;
  const float* __restrict__ d = x + (size_t)row * 2 * XN;

  __shared__ int smem[256 * 8];
  __shared__ int scnt[256];

  int loc[8];
  int nloc = 0, cnt = 0;

  auto proc = [&](long i, float l, float r) {
    if (i < (long)(XN - 1)) {
      bool disc = fabsf(l - r) > THRF;
      cnt += disc ? 1 : 0;
      if (disc && nloc < 8) loc[nloc++] = (int)i;
    }
  };

  f32x4 v = __builtin_nontemporal_load((const f32x4*)(d + base));
  proc(base + 0, v[0], v[1]);
  proc(base + 1, v[1], v[2]);
  proc(base + 2, v[2], v[3]);
  float carry = v[3];
  for (int t = 1; t < S / 4; ++t) {
    f32x4 w = __builtin_nontemporal_load((const f32x4*)(d + base + 4 * t));
    proc(base + 4 * t - 1, carry, w[0]);
    proc(base + 4 * t + 0, w[0], w[1]);
    proc(base + 4 * t + 1, w[1], w[2]);
    proc(base + 4 * t + 2, w[2], w[3]);
    carry = w[3];
  }
  if (base + S < XN) proc(base + S - 1, carry, d[base + S]);

#pragma unroll
  for (int j = 0; j < 8; ++j) smem[tid * 8 + j] = (j < nloc) ? loc[j] : 0x7fffffff;
  scnt[tid] = cnt;
  __syncthreads();

  if (tid == 0) {
    int total = 0;
    for (int t = 0; t < 256; ++t) total += scnt[t];
    cntpart[row * 8 + chunk] = total;
    int out8[8];
    int n = 0;
    for (int t = 0; t < 2048 && n < 8; ++t) {
      int id = smem[t];
      if (id != 0x7fffffff) out8[n++] = id;
    }
    for (int j = 0; j < 8; ++j)
      cand[(row * 8 + chunk) * 8 + j] = (j < n) ? out8[j] : 0x7fffffff;
  }
}

// ---------------------------------------------------------------------------
// Kernel W: one-time f32 -> f16 transpose of MLP weights into workspace.
// Layouts: W1t[i][n=512][k=256], W2t[i][n=256][k=512], W3t[n=128][k=256]
// (N-major, K contiguous => WMMA B-fragments are two contiguous v8h loads)
// ---------------------------------------------------------------------------
__global__ __launch_bounds__(256) void prep_weights(const float* __restrict__ rb_w1,
                                                    const float* __restrict__ rb_w2,
                                                    const float* __restrict__ out_w1,
                                                    _Float16* __restrict__ W1t,
                                                    _Float16* __restrict__ W2t,
                                                    _Float16* __restrict__ W3t) {
  const int N1  = 3 * 512 * 256;
  int gid = blockIdx.x * blockDim.x + threadIdx.x;
  if (gid < N1) {
    int i = gid / (512 * 256), r = gid % (512 * 256);
    int n = r / 256, k = r % 256;
    W1t[gid] = (_Float16)rb_w1[(size_t)i * 256 * 512 + (size_t)k * 512 + n];
  } else if (gid < 2 * N1) {
    int g = gid - N1;
    int i = g / (256 * 512), r = g % (256 * 512);
    int n = r / 512, k = r % 512;
    W2t[g] = (_Float16)rb_w2[(size_t)i * 512 * 256 + (size_t)k * 256 + n];
  } else if (gid < 2 * N1 + 128 * 256) {
    int g = gid - 2 * N1;
    int n = g / 256, k = g % 256;
    W3t[g] = (_Float16)out_w1[(size_t)k * 128 + n];
  }
}

// ---------------------------------------------------------------------------
// Kernel B: merge per-chunk candidates (already globally ordered), gather
// uL/uR/fc/valid, build 6-wide features, write channels 2..5 + front_count.
// ---------------------------------------------------------------------------
__global__ void select_gather(const float* __restrict__ x,
                              const int* __restrict__ cntpart,
                              const int* __restrict__ cand,
                              float* __restrict__ feats,
                              float* __restrict__ out) {
  const int row = blockIdx.x;
  const int tid = threadIdx.x;
  __shared__ int idx[8];
  if (tid == 0) {
    int total = 0;
    for (int c = 0; c < 8; ++c) total += cntpart[row * 8 + c];
    int n = 0;
    for (int t = 0; t < 64 && n < 8; ++t) {
      int v = cand[row * 64 + t];
      if (v != 0x7fffffff) idx[n++] = v;
    }
    for (; n < 8; ++n) idx[n] = 0;
    out[NB * 6 * 8 + row] = (float)total;  // front_count appended after front_params
  }
  __syncthreads();
  if (tid < 8) {
    int i = idx[tid];
    const float* d = x + (size_t)row * 2 * XN;
    const float* c = d + XN;
    float uL = d[i], uR = d[i + 1];
    float fc = 0.5f * (c[i] + c[i + 1]);
    float diff = uL - uR;
    float valid = (fabsf(diff) > THRF) ? 1.0f : 0.0f;
    float* f = feats + (size_t)(row * 8 + tid) * 6;
    f[0] = uL;
    f[1] = uR;
    f[2] = diff;
    f[3] = fabsf(diff);
    f[4] = 0.5f * (uL + uR);
    f[5] = (diff > 0.f) ? 1.f : ((diff < 0.f) ? -1.f : 0.f);
    out[(row * 6 + 2) * 8 + tid] = uL;
    out[(row * 6 + 3) * 8 + tid] = uR;
    out[(row * 6 + 4) * 8 + tid] = fc;
    out[(row * 6 + 5) * 8 + tid] = valid;
  }
}

// ---------------------------------------------------------------------------
// WMMA fragment loaders (wave32 16x16x32 f16 layouts per CDNA5 ISA 7.12.2):
// lane L holds row (A) / col (B) = L%16; lanes 0-15 carry K = kb..kb+7 and
// kb+16..kb+23 with kb = 32*kstep + 8*(L>=16); halves packed v[0..7],v[8..15].
// ---------------------------------------------------------------------------
__device__ __forceinline__ v16h load_a_frag(const _Float16* As, int lda, int ks) {
  const int lane = threadIdx.x & 31;
  const int m = lane & 15;
  const int kb = ks * 32 + ((lane & 16) ? 8 : 0);
  const _Float16* p = As + (size_t)m * lda + kb;
  v8h lo = *(const v8h*)p;
  v8h hi = *(const v8h*)(p + 16);
  v16h a;
#pragma unroll
  for (int j = 0; j < 8; ++j) { a[j] = lo[j]; a[8 + j] = hi[j]; }
  return a;
}

__device__ __forceinline__ v16h load_b_frag(const _Float16* Wt, int ldk, int ncol, int ks) {
  const int lane = threadIdx.x & 31;
  const int kb = ks * 32 + ((lane & 16) ? 8 : 0);
  const _Float16* p = Wt + (size_t)(ncol + (lane & 15)) * ldk + kb;
  v8h lo = *(const v8h*)p;
  v8h hi = *(const v8h*)(p + 16);
  v16h b;
#pragma unroll
  for (int j = 0; j < 8; ++j) { b[j] = lo[j]; b[8 + j] = hi[j]; }
  return b;
}

__device__ __forceinline__ float gelu_f(float v) {
  return 0.5f * v * (1.0f + erff(v * 0.70710678118654752f));
}

// LayerNorm over H=256 for 16 rows in LDS; wave32 shuffle reduction.
// dogelu=true: in-place h = gelu(LN(h)).  else: As = (f16)LN(h), Hs untouched.
__device__ __forceinline__ void ln_stage(float (*Hs)[HH], const float* g, const float* bt,
                                         int wave, int lane, bool dogelu, _Float16* dstA) {
#pragma unroll
  for (int rr = 0; rr < 2; ++rr) {
    const int m = wave * 2 + rr;
    float xv[8], s = 0.f, s2 = 0.f;
#pragma unroll
    for (int t = 0; t < 8; ++t) {
      float v = Hs[m][lane + 32 * t];
      xv[t] = v; s += v; s2 += v * v;
    }
#pragma unroll
    for (int off = 16; off > 0; off >>= 1) {
      s  += __shfl_xor(s,  off, 32);
      s2 += __shfl_xor(s2, off, 32);
    }
    const float mu = s * (1.0f / HH);
    const float var = s2 * (1.0f / HH) - mu * mu;
    const float rs = rsqrtf(var + 1e-5f);
#pragma unroll
    for (int t = 0; t < 8; ++t) {
      const int c = lane + 32 * t;
      float v = (xv[t] - mu) * rs * g[c] + bt[c];
      if (dogelu) Hs[m][c] = gelu_f(v);
      else        dstA[(size_t)m * HH + c] = (_Float16)v;
    }
  }
}

// ---------------------------------------------------------------------------
// Kernel C: 512-sample MLP. 32 blocks x 16 rows, 256 threads = 8 wave32.
// All GEMMs via v_wmma_f32_16x16x32_f16; residual stream in LDS f32.
// ---------------------------------------------------------------------------
__global__ __launch_bounds__(256) void mlp_kernel(
    const float* __restrict__ feats, const float* __restrict__ w_in,
    const float* __restrict__ b_in, const float* __restrict__ ln_in_g,
    const float* __restrict__ ln_in_b, const float* __restrict__ rb_ln_g,
    const float* __restrict__ rb_ln_b, const _Float16* __restrict__ W1t,
    const float* __restrict__ rb_b1, const _Float16* __restrict__ W2t,
    const float* __restrict__ rb_b2, const float* __restrict__ out_ln_g,
    const float* __restrict__ out_ln_b, const _Float16* __restrict__ W3t,
    const float* __restrict__ out_b1, const float* __restrict__ out_w2,
    float* __restrict__ out) {
  __shared__ float    Hs[16][HH];     // residual stream, f32 (16 KB)
  __shared__ _Float16 As[16][HH];     // LN output, WMMA A operand (8 KB)
  __shared__ _Float16 Zs[16][512];    // hidden activations / f32 head scratch (16 KB)

  const int tid  = threadIdx.x;
  const int lane = tid & 31;
  const int wave = tid >> 5;
  const int r0   = blockIdx.x * 16;
  const int mb   = (lane & 16) ? 8 : 0;  // C/D layout: VGPR v -> row v + mb

  // ---- input projection: lin = feats @ w_in + b_in (K=6, scalar FMA) ----
  {
    float wcol[6];
#pragma unroll
    for (int j = 0; j < 6; ++j) wcol[j] = w_in[j * HH + tid];
    const float bb = b_in[tid];
    for (int m = 0; m < 16; ++m) {
      const float* f = feats + (size_t)(r0 + m) * 6;
      float acc = bb;
#pragma unroll
      for (int j = 0; j < 6; ++j) acc += f[j] * wcol[j];
      Hs[m][tid] = acc;
    }
  }
  __syncthreads();
  ln_stage(Hs, ln_in_g, ln_in_b, wave, lane, true, nullptr);  // h = gelu(LN(lin))
  __syncthreads();

  const v8f vzero = {0.f, 0.f, 0.f, 0.f, 0.f, 0.f, 0.f, 0.f};

  // ---- 3 residual blocks ----
  for (int blk = 0; blk < 3; ++blk) {
    ln_stage(Hs, rb_ln_g + blk * HH, rb_ln_b + blk * HH, wave, lane, false, &As[0][0]);
    __syncthreads();

    // GEMM1: Z(16x512) = As(16x256) @ W1t  — 8 waves x 4 N-tiles, 8 K-steps
    const _Float16* W1 = W1t + (size_t)blk * 512 * 256;
    v8f acc1[4] = {vzero, vzero, vzero, vzero};
    for (int ks = 0; ks < 8; ++ks) {
      v16h a = load_a_frag(&As[0][0], HH, ks);
#pragma unroll
      for (int jt = 0; jt < 4; ++jt) {
        v16h b = load_b_frag(W1, HH, (wave * 4 + jt) * 16, ks);
        acc1[jt] = __builtin_amdgcn_wmma_f32_16x16x32_f16(
            false, a, false, b, (short)0, acc1[jt], false, false);
      }
    }
#pragma unroll
    for (int jt = 0; jt < 4; ++jt) {
      const int n = (wave * 4 + jt) * 16 + (lane & 15);
      const float bias = rb_b1[blk * 512 + n];
#pragma unroll
      for (int v = 0; v < 8; ++v)
        Zs[mb + v][n] = (_Float16)gelu_f(acc1[jt][v] + bias);
    }
    __syncthreads();

    // GEMM2: Y(16x256) = Zs(16x512) @ W2t — 8 waves x 2 N-tiles, 16 K-steps
    const _Float16* W2 = W2t + (size_t)blk * 256 * 512;
    v8f acc2[2] = {vzero, vzero};
    for (int ks = 0; ks < 16; ++ks) {
      v16h a = load_a_frag(&Zs[0][0], 512, ks);
#pragma unroll
      for (int jt = 0; jt < 2; ++jt) {
        v16h b = load_b_frag(W2, 512, (wave * 2 + jt) * 16, ks);
        acc2[jt] = __builtin_amdgcn_wmma_f32_16x16x32_f16(
            false, a, false, b, (short)0, acc2[jt], false, false);
      }
    }
#pragma unroll
    for (int jt = 0; jt < 2; ++jt) {
      const int n = (wave * 2 + jt) * 16 + (lane & 15);
      const float bias = rb_b2[blk * HH + n];
#pragma unroll
      for (int v = 0; v < 8; ++v) Hs[mb + v][n] += acc2[jt][v] + bias;
    }
    __syncthreads();
  }

  // ---- output head ----
  ln_stage(Hs, out_ln_g, out_ln_b, wave, lane, false, &As[0][0]);
  __syncthreads();

  v8f acc3 = vzero;  // 16x128 = As(16x256) @ W3t — one N-tile per wave
  for (int ks = 0; ks < 8; ++ks) {
    v16h a = load_a_frag(&As[0][0], HH, ks);
    v16h b = load_b_frag(W3t, HH, wave * 16, ks);
    acc3 = __builtin_amdgcn_wmma_f32_16x16x32_f16(
        false, a, false, b, (short)0, acc3, false, false);
  }
  float* Gf = (float*)&Zs[0][0];  // reuse as 16x128 f32
  {
    const int n = wave * 16 + (lane & 15);
    const float bias = out_b1[n];
#pragma unroll
    for (int v = 0; v < 8; ++v)
      Gf[(size_t)(mb + v) * 128 + n] = gelu_f(acc3[v] + bias);
  }
  __syncthreads();

  // final 128 -> 2 projection; 32 dot products handled by wave 0
  if (tid < 32) {
    const int m = tid >> 1, j = tid & 1;
    float s = 0.f;
    for (int k = 0; k < 128; ++k) s += Gf[(size_t)m * 128 + k] * out_w2[k * 2 + j];
    const int smp = r0 + m;
    const int b = smp >> 3, kk = smp & 7;
    out[(b * 6 + j) * 8 + kk] = s;
  }
}

// ---------------------------------------------------------------------------
extern "C" void kernel_launch(void* const* d_in, const int* in_sizes, int n_in,
                              void* d_out, int out_size, void* d_ws, size_t ws_size,
                              hipStream_t stream) {
  const float* x        = (const float*)d_in[0];
  const float* w_in     = (const float*)d_in[1];
  const float* b_in     = (const float*)d_in[2];
  const float* ln_in_g  = (const float*)d_in[3];
  const float* ln_in_b  = (const float*)d_in[4];
  const float* rb_ln_g  = (const float*)d_in[5];
  const float* rb_ln_b  = (const float*)d_in[6];
  const float* rb_w1    = (const float*)d_in[7];
  const float* rb_b1    = (const float*)d_in[8];
  const float* rb_w2    = (const float*)d_in[9];
  const float* rb_b2    = (const float*)d_in[10];
  const float* out_ln_g = (const float*)d_in[11];
  const float* out_ln_b = (const float*)d_in[12];
  const float* out_w1   = (const float*)d_in[13];
  const float* out_b1   = (const float*)d_in[14];
  const float* out_w2   = (const float*)d_in[15];
  float* out = (float*)d_out;

  // workspace carve (all chunks 256B-aligned)
  char* w = (char*)d_ws;
  int* cntpart = (int*)w;        w += 64 * 8 * 4;          // 2 KB
  int* cand    = (int*)w;        w += 64 * 8 * 8 * 4;      // 16 KB
  float* feats = (float*)w;      w += 512 * 6 * 4;         // 12 KB
  _Float16* W1t = (_Float16*)w;  w += (size_t)3 * 512 * 256 * 2;  // 768 KB
  _Float16* W2t = (_Float16*)w;  w += (size_t)3 * 256 * 512 * 2;  // 768 KB
  _Float16* W3t = (_Float16*)w;  w += (size_t)128 * 256 * 2;      // 64 KB

  dim3 gScan(8, NB);
  scan_fronts<<<gScan, 256, 0, stream>>>(x, cntpart, cand);
  prep_weights<<<3200, 256, 0, stream>>>(rb_w1, rb_w2, out_w1, W1t, W2t, W3t);
  select_gather<<<NB, 64, 0, stream>>>(x, cntpart, cand, feats, out);
  mlp_kernel<<<32, 256, 0, stream>>>(feats, w_in, b_in, ln_in_g, ln_in_b,
                                     rb_ln_g, rb_ln_b, W1t, rb_b1, W2t, rb_b2,
                                     out_ln_g, out_ln_b, W3t, out_b1, out_w2, out);
}
